// Dyan_22728966931067
// MI455X (gfx1250) — matrix operand order repaired and checked by reading, not compiled, not run on hardware
//
#include <hip/hip_runtime.h>
#include <stdint.h>

#define T36     36
#define NPDIM   161
#define MDIM    50
#define LAMBDA_F 0.005f
#define MAX_ITER 100

// ws layout (float offsets)
#define WS_D   0        // 36*161 = 5796 floats
#define WS_A   6144     // 161*161 = 25921 floats (DtD, then A in place)
#define WS_SC  32128    // [0]=Linv, [1]=wl
#define WS_TT  32192    // 100 floats: momentum coefficients tt[it]

#define YPITCH 100               // dwords per column (bank-spread padding)
#define BUFSZ  (64 * YPITCH)     // one hi or lo array: 6400 dwords

typedef __attribute__((ext_vector_type(16))) __bf16    v16bf;
typedef __attribute__((ext_vector_type(8)))  float     v8f;
typedef __attribute__((ext_vector_type(8)))  uint32_t  v8u;

static __device__ __forceinline__ uint32_t f2bf(float f) {
  uint32_t u = __float_as_uint(f);
  return (u + 0x7fffu + ((u >> 16) & 1u)) >> 16;   // RNE truncate to bf16
}
static __device__ __forceinline__ float bf2f(uint32_t h) {
  return __uint_as_float(h << 16);
}

// ---------------- Kernel 1: build normalized dictionary D [36][161] ----------
__global__ void build_dict_kernel(const float* __restrict__ r,
                                  const float* __restrict__ theta,
                                  float* __restrict__ ws) {
  int p = threadIdx.x;
  if (p >= NPDIM) return;
  float* D = ws + WS_D;
  float rr = 0.f, th = 0.f;
  if (p >= 1 && p <= 80)  { rr = r[p - 1];  th = theta[p - 1]; }
  if (p >= 81)            { rr = r[p - 81]; th = theta[p - 81]; }
  float ss = 0.f;
  for (int i = 0; i < T36; ++i) {
    float fi = (float)i, v;
    if (p == 0)       v = 1.0f + 1e-10f;
    else if (p <= 80) v = powf(rr, fi) * cosf(fi * th);
    else              v = powf(rr, fi) * sinf(fi * th);
    ss += v * v;
  }
  float inv = 1.0f / sqrtf(ss);
  for (int i = 0; i < T36; ++i) {
    float fi = (float)i, v;
    if (p == 0)       v = 1.0f + 1e-10f;
    else if (p <= 80) v = powf(rr, fi) * cosf(fi * th);
    else              v = powf(rr, fi) * sinf(fi * th);
    D[i * NPDIM + p] = v * inv;
  }
}

// ---- Kernel 2: DtD, spectral norm (power iter), A = I - DtD/L, tt table ----
__global__ __launch_bounds__(256) void build_A_kernel(float* __restrict__ ws) {
  const float* D = ws + WS_D;
  float* A = ws + WS_A;
  const int tid = threadIdx.x;

  for (int idx = tid; idx < NPDIM * NPDIM; idx += 256) {
    int pp = idx / NPDIM, q = idx - pp * NPDIM;
    float s = 0.f;
    for (int t = 0; t < T36; ++t) s += D[t * NPDIM + pp] * D[t * NPDIM + q];
    A[idx] = s;                         // DtD (temporarily)
  }
  __threadfence_block();
  __syncthreads();

  __shared__ float v[NPDIM], red[NPDIM], nrm_s;
  if (tid < NPDIM) v[tid] = 1.0f;
  __syncthreads();
  for (int it = 0; it < 150; ++it) {    // power iteration on PSD Gram matrix
    float wv = 0.f;
    if (tid < NPDIM) {
      for (int q = 0; q < NPDIM; ++q) wv += A[tid * NPDIM + q] * v[q];
      red[tid] = wv * wv;
    }
    __syncthreads();
    if (tid == 0) {
      float s = 0.f;
      for (int i = 0; i < NPDIM; ++i) s += red[i];
      nrm_s = sqrtf(s);
    }
    __syncthreads();
    if (tid < NPDIM) v[tid] = wv / nrm_s;
    __syncthreads();
  }
  float Linv = 1.0f / nrm_s;
  if (tid == 0) {
    ws[WS_SC]     = Linv;
    ws[WS_SC + 1] = LAMBDA_F * Linv;
    // precompute FISTA momentum coefficients tt[it] = (t_old-1)/t_new
    float t_old = 1.0f;
    for (int it = 0; it < MAX_ITER; ++it) {
      float t_new = 0.5f * (1.0f + sqrtf(1.0f + 4.0f * t_old * t_old));
      ws[WS_TT + it] = (t_old - 1.0f) / t_new;
      t_old = t_new;
    }
  }
  for (int idx = tid; idx < NPDIM * NPDIM; idx += 256) {
    int pp = idx / NPDIM, q = idx - pp * NPDIM;
    A[idx] = ((pp == q) ? 1.0f : 0.0f) - A[idx] * Linv;
  }
}

// ---------------- Kernel 3: per-sample FISTA, state resident on-chip --------
__global__ __launch_bounds__(352, 1)
void fista_kernel(const float* __restrict__ yin, const float* __restrict__ ws,
                  float* __restrict__ out, int N) {
  // smem: [0, 4*BUFSZ) = y-state: buf0_hi, buf0_lo, buf1_hi, buf1_lo (packed bf16 pairs)
  //       [4*BUFSZ, +176*64) = c = DtY * Linv  (fp32)
  __shared__ uint32_t smem[4 * BUFSZ + 176 * 64];
  uint32_t* ypk  = smem;
  float*    cbuf = (float*)(smem + 4 * BUFSZ);

  const int n    = blockIdx.x;
  const int tid  = threadIdx.x;
  const int pt   = tid >> 5;            // wave id == row-tile (0..10)
  const int lane = tid & 31;
  const int l15  = lane & 15;
  const bool hiH = (lane >= 16);

  const float* Dw   = ws + WS_D;
  const float* Aw   = ws + WS_A;
  const float* tts  = ws + WS_TT;
  const float  Linv = ws[WS_SC];
  const float  wl   = ws[WS_SC + 1];
  const float* Yn   = yin + (size_t)n * (T36 * MDIM);

  // Phase 1: stage Y_n (aliases ypk region, zeroed afterwards)
  float* Ys = (float*)smem;
  for (int i = tid; i < T36 * MDIM; i += 352) Ys[i] = Yn[i];
  __syncthreads();

  // Phase 2: c[p][m] = Linv * sum_t D[t][p] * Y[t][m]  (0 in padding)
  for (int idx = tid; idx < 176 * 64; idx += 352) {
    int p = idx >> 6, m = idx & 63;
    float s = 0.f;
    if (p < NPDIM && m < MDIM) {
      for (int t = 0; t < T36; ++t) s += Dw[t * NPDIM + p] * Ys[t * MDIM + m];
      s *= Linv;
    }
    cbuf[idx] = s;
  }
  __syncthreads();

  // Phase 3: zero both y-state buffers (y0 = 0; pad rows stay 0 forever)
  for (int i = tid; i < 4 * BUFSZ; i += 352) ypk[i] = 0u;

  // Phase 4: preload A fragments into registers (hi/lo bf16 split), reused 100x
  const int m_idx = pt * 16 + l15;          // A-frag row (M)
  const int kadd  = hiH ? 16 : 0;
  v16bf ahi[6], alo[6];
#pragma unroll
  for (int ks = 0; ks < 6; ++ks) {
    v8u uh, ul;
#pragma unroll
    for (int v = 0; v < 8; ++v) {
      int kb = ks * 32 + kadd + 2 * v;
      float a0 = 0.f, a1 = 0.f;
      if (m_idx < NPDIM) {
        if (kb < NPDIM)     a0 = Aw[m_idx * NPDIM + kb];
        if (kb + 1 < NPDIM) a1 = Aw[m_idx * NPDIM + kb + 1];
      }
      uint32_t h0 = f2bf(a0), h1 = f2bf(a1);
      uint32_t l0 = f2bf(a0 - bf2f(h0)), l1 = f2bf(a1 - bf2f(h1));
      uh[v] = h0 | (h1 << 16);
      ul[v] = l0 | (l1 << 16);
    }
    ahi[ks] = __builtin_bit_cast(v16bf, uh);
    alo[ks] = __builtin_bit_cast(v16bf, ul);
  }

  float xold[4][8];
#pragma unroll
  for (int nt = 0; nt < 4; ++nt)
#pragma unroll
    for (int r = 0; r < 8; ++r) xold[nt][r] = 0.f;

  __syncthreads();

  // -------- FISTA main loop --------
  const int half8 = hiH ? 8 : 0;
  const int crow  = pt * 16 + half8;        // first C/D row held by this lane
  int rb = 0;
  for (int it = 0; it < MAX_ITER; ++it) {
    const float tt = tts[it];               // uniform scalar load (precomputed)
    int wb = rb ^ 1;
    const uint32_t* bh = ypk + rb * 2 * BUFSZ;
    const uint32_t* bl = bh + BUFSZ;
    uint32_t* wh = ypk + wb * 2 * BUFSZ;
    uint32_t* wlp = wh + BUFSZ;

#pragma unroll
    for (int nt = 0; nt < 4; ++nt) {
      const int ncol = nt * 16 + l15;
      v8f acc = {0.f, 0.f, 0.f, 0.f, 0.f, 0.f, 0.f, 0.f};
#pragma unroll
      for (int ks = 0; ks < 6; ++ks) {
        int base = ncol * YPITCH + ks * 16 + (hiH ? 8 : 0);
        uint4 h0 = *(const uint4*)(bh + base);
        uint4 h1 = *(const uint4*)(bh + base + 4);
        uint4 g0 = *(const uint4*)(bl + base);
        uint4 g1 = *(const uint4*)(bl + base + 4);
        v8u uhh = {h0.x, h0.y, h0.z, h0.w, h1.x, h1.y, h1.z, h1.w};
        v8u ull = {g0.x, g0.y, g0.z, g0.w, g1.x, g1.y, g1.z, g1.w};
        v16bf bhi = __builtin_bit_cast(v16bf, uhh);
        v16bf blo = __builtin_bit_cast(v16bf, ull);
        acc = __builtin_amdgcn_wmma_f32_16x16x32_bf16(false, ahi[ks], false, bhi,
                                                      (short)0, acc, false, false);
        acc = __builtin_amdgcn_wmma_f32_16x16x32_bf16(false, ahi[ks], false, blo,
                                                      (short)0, acc, false, false);
        acc = __builtin_amdgcn_wmma_f32_16x16x32_bf16(false, alo[ks], false, bhi,
                                                      (short)0, acc, false, false);
      }
      // soft-threshold + momentum, pack y_new as bf16 hi/lo pairs
      float yv[8];
#pragma unroll
      for (int r = 0; r < 8; ++r) {
        float s = acc[r] + cbuf[(crow + r) * 64 + ncol];
        float xn = fmaxf(0.f, s - wl) + fminf(0.f, s + wl);
        yv[r] = xn + tt * (xn - xold[nt][r]);
        xold[nt][r] = xn;
      }
      uint32_t ph[4], pl[4];
#pragma unroll
      for (int j = 0; j < 4; ++j) {
        float ye = yv[2 * j], yo = yv[2 * j + 1];
        uint32_t he = f2bf(ye), ho = f2bf(yo);
        uint32_t le = f2bf(ye - bf2f(he)), lo = f2bf(yo - bf2f(ho));
        ph[j] = he | (ho << 16);
        pl[j] = le | (lo << 16);
      }
      int wbase = ncol * YPITCH + pt * 8 + (hiH ? 4 : 0);
      *(uint4*)(wh + wbase)  = make_uint4(ph[0], ph[1], ph[2], ph[3]);
      *(uint4*)(wlp + wbase) = make_uint4(pl[0], pl[1], pl[2], pl[3]);
    }
    __syncthreads();
    rb = wb;
  }

  // -------- write C_pred --------
  float* outC = out + (size_t)N * (T36 * MDIM) + (size_t)n * (NPDIM * MDIM);
#pragma unroll
  for (int nt = 0; nt < 4; ++nt) {
    int m = nt * 16 + l15;
#pragma unroll
    for (int r = 0; r < 8; ++r) {
      int p = crow + r;
      if (p < NPDIM && m < MDIM) outC[p * MDIM + m] = xold[nt][r];
    }
  }

  // -------- Y_pred = D @ C (stage x into LDS, aliasing y-state region) ------
  float* xl = (float*)smem;                 // [176][64] fp32, 11264 dwords
#pragma unroll
  for (int nt = 0; nt < 4; ++nt)
#pragma unroll
    for (int r = 0; r < 8; ++r)
      xl[(crow + r) * 64 + (nt * 16 + l15)] = xold[nt][r];
  __syncthreads();

  float* outY = out + (size_t)n * (T36 * MDIM);
  for (int idx = tid; idx < T36 * MDIM; idx += 352) {
    int t = idx / MDIM, m = idx - t * MDIM;
    float s = 0.f;
    for (int p = 0; p < NPDIM; ++p) s += Dw[t * NPDIM + p] * xl[p * 64 + m];
    outY[idx] = s;
  }
}

extern "C" void kernel_launch(void* const* d_in, const int* in_sizes, int n_in,
                              void* d_out, int out_size, void* d_ws, size_t ws_size,
                              hipStream_t stream) {
  const float* y     = (const float*)d_in[0];
  const float* r     = (const float*)d_in[1];
  const float* theta = (const float*)d_in[2];
  float* out = (float*)d_out;
  float* ws  = (float*)d_ws;
  int N = in_sizes[0] / (T36 * MDIM);   // 4096

  build_dict_kernel<<<1, 192, 0, stream>>>(r, theta, ws);
  build_A_kernel<<<1, 256, 0, stream>>>(ws);
  fista_kernel<<<N, 352, 0, stream>>>(y, ws, out, N);
}
